// selfattention_11811160064128
// MI455X (gfx1250) — compile-verified, matrix operand order checked by reading, not run on hardware
//
#include <hip/hip_runtime.h>
#include <hip/hip_bf16.h>

typedef __attribute__((ext_vector_type(16))) __bf16 v16bf;
typedef __attribute__((ext_vector_type(8)))  __bf16 v8bf;
typedef __attribute__((ext_vector_type(8)))  float  v8f;
typedef __attribute__((ext_vector_type(4)))  int    v4i;
typedef __attribute__((address_space(1))) v4i gv4i;   // global b128
typedef __attribute__((address_space(3))) v4i lv4i;   // LDS b128

#define BATCH  4
#define SEQ    2048
#define CH     1024
#define NHEADS 16
#define HD     64

// ---------- CDNA5 async global->LDS path (guarded; fallback = sync copy) ----
#if defined(__has_builtin)
#  if __has_builtin(__builtin_amdgcn_global_load_async_to_lds_b128)
#    define HAVE_ASYNC 1
#  endif
#endif
#ifndef HAVE_ASYNC
#  define HAVE_ASYNC 0
#endif

__device__ __forceinline__ void cp_async_b128(const __bf16* g, __bf16* l) {
#if HAVE_ASYNC
  __builtin_amdgcn_global_load_async_to_lds_b128(
      (gv4i*)(unsigned long long)(g),
      (lv4i*)(unsigned)(unsigned long long)(l),
      0, 0);
#else
  *(v8bf*)l = *(const v8bf*)g;
#endif
}

#if HAVE_ASYNC
#  if __has_builtin(__builtin_amdgcn_s_wait_asynccnt)
#    define WAIT_ASYNC(n) __builtin_amdgcn_s_wait_asynccnt(n)
#  else
#    define WAIT_ASYNC(n) asm volatile("s_wait_asynccnt %0" :: "i"(n) : "memory")
#  endif
#else
#  define WAIT_ASYNC(n) do {} while (0)
#endif

// ---------------------------------------------------------------------------
__device__ __forceinline__ __bf16 f2bf(float f) {
  unsigned u = __builtin_bit_cast(unsigned, f);
  u += 0x7FFFu + ((u >> 16) & 1u);           // round-to-nearest-even
  unsigned short h = (unsigned short)(u >> 16);
  return __builtin_bit_cast(__bf16, h);
}

__device__ __forceinline__ v8f wmma_bf16(v16bf a, v16bf b, v8f c) {
  return __builtin_amdgcn_wmma_f32_16x16x32_bf16(false, a, false, b, (short)0, c, false, false);
}

// 16x32 bf16 A/B fragment from row-major matrix (works for global or LDS).
// lane m -> K{0..7,16..23}, lane m+16 -> K{8..15,24..31}.
__device__ __forceinline__ v16bf load_frag(const __bf16* p, int ld, int r0, int k0) {
  int lane = threadIdx.x & 31;
  int r  = r0 + (lane & 15);
  int ks = k0 + ((lane & 16) ? 8 : 0);
  const v8bf* q = (const v8bf*)(p + (long)r * ld + ks);
  v8bf lo = q[0];
  v8bf hi = q[2];          // +16 elements = +32B
  v16bf out;
#pragma unroll
  for (int i = 0; i < 8; i++) { out[i] = lo[i]; out[i + 8] = hi[i]; }
  return out;
}

__global__ void convert_f32_bf16(const float* __restrict__ src, __bf16* __restrict__ dst, int n) {
  int i = blockIdx.x * blockDim.x + threadIdx.x;
  int stride = gridDim.x * blockDim.x;
  for (; i < n; i += stride) dst[i] = f2bf(src[i]);
}

// QKV projection: [8192,1024] @ [3072,1024]^T, 64x64 tile per wave (16 WMMA / 32-K step)
__global__ void __launch_bounds__(128) qkv_gemm(const __bf16* __restrict__ X,
                                                const __bf16* __restrict__ W,
                                                __bf16* __restrict__ qb,
                                                __bf16* __restrict__ kb,
                                                __bf16* __restrict__ vt) {
  int wave = threadIdx.x >> 5;
  int lane = threadIdx.x & 31;
  int n0 = blockIdx.x * 256 + wave * 64;
  int m0 = blockIdx.y * 64;
  v8f acc[4][4] = {};
  for (int k0 = 0; k0 < CH; k0 += 32) {
    v16bf a[4], b[4];
#pragma unroll
    for (int i = 0; i < 4; i++) a[i] = load_frag(X, CH, m0 + i * 16, k0);
#pragma unroll
    for (int j = 0; j < 4; j++) b[j] = load_frag(W, CH, n0 + j * 16, k0);
#pragma unroll
    for (int i = 0; i < 4; i++)
#pragma unroll
      for (int j = 0; j < 4; j++)
        acc[i][j] = wmma_bf16(a[i], b[j], acc[i][j]);
  }
  int half8 = (lane & 16) ? 8 : 0;
  int ncol  = lane & 15;
#pragma unroll
  for (int i = 0; i < 4; i++)
#pragma unroll
    for (int j = 0; j < 4; j++)
#pragma unroll
      for (int v = 0; v < 8; v++) {
        int m = m0 + i * 16 + v + half8;
        int n = n0 + j * 16 + ncol;
        int sec = n >> 10;                 // 0=Q 1=K 2=V
        int c = n & 1023;
        int h = c >> 6, d = c & 63;
        int b2 = m >> 11, l = m & 2047;
        long bh = (long)(b2 * NHEADS + h);
        __bf16 bv = f2bf(acc[i][j][v]);
        if (sec == 0)      qb[(bh * SEQ + l) * HD + d] = bv;
        else if (sec == 1) kb[(bh * SEQ + l) * HD + d] = bv;
        else               vt[(bh * HD + d) * SEQ + l] = bv;
      }
}

// Flash attention: one wave per (b,h, 32-query tile) = two 16-row Q tiles
// sharing each async-staged, double-buffered 32-key K/V LDS tile.
#define KPAD 72   // 32-row K tile, 64+8 pad elems/row  -> conflict-free ds_load_b128
#define VPAD 40   // 64-row V^T tile, 32+8 pad elems/row
__global__ void __launch_bounds__(32) attn_fwd(const __bf16* __restrict__ qb,
                                               const __bf16* __restrict__ kb,
                                               const __bf16* __restrict__ vt,
                                               __bf16* __restrict__ ctx) {
  __shared__ __align__(16) __bf16 kst[2][32 * KPAD];
  __shared__ __align__(16) __bf16 vst[2][64 * VPAD];
  __shared__ __align__(16) __bf16 pshare[2][16 * 32];
  int lane = threadIdx.x & 31;
  int q0 = blockIdx.x * 32;
  int bh = blockIdx.y;
  const __bf16* Q  = qb + (long)bh * SEQ * HD;
  const __bf16* K  = kb + (long)bh * SEQ * HD;
  const __bf16* VT = vt + (long)bh * HD * SEQ;

  v16bf aq[2][2];
#pragma unroll
  for (int u = 0; u < 2; u++) {
    aq[u][0] = load_frag(Q, HD, q0 + u * 16, 0);
    aq[u][1] = load_frag(Q, HD, q0 + u * 16, 32);
  }

  v8f o[2][4] = {};
  float mrow[2][8], lrow[2][8];
#pragma unroll
  for (int u = 0; u < 2; u++)
#pragma unroll
    for (int v = 0; v < 8; v++) { mrow[u][v] = -__builtin_inff(); lrow[u][v] = 0.f; }

  const float scale = 0.125f;               // 1/sqrt(64)
  int half8 = (lane & 16) ? 8 : 0;
  int ncol  = lane & 15;
  int kend = q0 + 32;                        // multiple of 32
  int ntiles = kend >> 5;

  // stage one 32-key tile: K[kt..kt+31][0..63] and V^T[0..63][kt..kt+31]
  auto stage = [&](int buf, int kt) {
#pragma unroll
    for (int c = 0; c < 8; c++) {           // K tile: 8 chunks/row-of-64
      int chunk = c * 32 + lane;
      int row = chunk >> 3, col = (chunk & 7) * 8;
      cp_async_b128(K + (long)(kt + row) * HD + col, &kst[buf][row * KPAD + col]);
    }
#pragma unroll
    for (int c = 0; c < 8; c++) {           // V^T tile: 4 chunks/row-of-32
      int chunk = c * 32 + lane;
      int row = chunk >> 2, col = (chunk & 3) * 8;
      cp_async_b128(VT + (long)row * SEQ + kt + col, &vst[buf][row * VPAD + col]);
    }
  };

  stage(0, 0);
  for (int t = 0; t < ntiles; t++) {
    int kt = t * 32;
    if (t + 1 < ntiles) { stage((t + 1) & 1, kt + 32); WAIT_ASYNC(16); }
    else                { WAIT_ASYNC(0); }
    const __bf16* Kt = &kst[t & 1][0];
    const __bf16* Vt = &vst[t & 1][0];

    v16bf bk[2][2];
#pragma unroll
    for (int nh = 0; nh < 2; nh++) {
      bk[nh][0] = load_frag(Kt, KPAD, nh * 16, 0);
      bk[nh][1] = load_frag(Kt, KPAD, nh * 16, 32);
    }
    v16bf bv[4];
#pragma unroll
    for (int j = 0; j < 4; j++) bv[j] = load_frag(Vt, VPAD, j * 16, 0);

#pragma unroll
    for (int u = 0; u < 2; u++) {
      v8f s[2];
#pragma unroll
      for (int nh = 0; nh < 2; nh++) {
        v8f tacc = {};
        tacc = wmma_bf16(aq[u][0], bk[nh][0], tacc);
        tacc = wmma_bf16(aq[u][1], bk[nh][1], tacc);
        s[nh] = tacc;
      }
      // scale + causal mask
#pragma unroll
      for (int nh = 0; nh < 2; nh++) {
        int col = kt + nh * 16 + ncol;
#pragma unroll
        for (int v = 0; v < 8; v++) {
          int row = q0 + u * 16 + v + half8;
          float val = s[nh][v] * scale;
          s[nh][v] = (col > row) ? -__builtin_inff() : val;
        }
      }
      // online softmax (reductions across the 16-lane half)
      float alpha[8];
#pragma unroll
      for (int v = 0; v < 8; v++) {
        float tm = fmaxf(s[0][v], s[1][v]);
#pragma unroll
        for (int off = 1; off < 16; off <<= 1) tm = fmaxf(tm, __shfl_xor(tm, off, 32));
        float mn = fmaxf(mrow[u][v], tm);
        alpha[v] = __expf(mrow[u][v] - mn);
        mrow[u][v] = mn;
      }
#pragma unroll
      for (int v = 0; v < 8; v++) {
        float p0 = __expf(s[0][v] - mrow[u][v]);
        float p1 = __expf(s[1][v] - mrow[u][v]);
        s[0][v] = p0; s[1][v] = p1;
        float ts = p0 + p1;
#pragma unroll
        for (int off = 1; off < 16; off <<= 1) ts += __shfl_xor(ts, off, 32);
        lrow[u][v] = lrow[u][v] * alpha[v] + ts;
      }
#pragma unroll
      for (int j = 0; j < 4; j++)
#pragma unroll
        for (int v = 0; v < 8; v++) o[u][j][v] *= alpha[v];

      // P (C-layout f32) -> LDS bf16 -> A-fragment layout
#pragma unroll
      for (int nh = 0; nh < 2; nh++)
#pragma unroll
        for (int v = 0; v < 8; v++)
          pshare[u][(v + half8) * 32 + nh * 16 + ncol] = f2bf(s[nh][v]);
      __syncthreads();
      v16bf ap;
      {
        int r  = lane & 15;
        int ks = (lane & 16) ? 8 : 0;
        const v8bf* lp = (const v8bf*)(&pshare[u][r * 32 + ks]);
        v8bf lo = lp[0], hi = lp[2];
#pragma unroll
        for (int i = 0; i < 8; i++) { ap[i] = lo[i]; ap[i + 8] = hi[i]; }
      }
      __syncthreads();
#pragma unroll
      for (int j = 0; j < 4; j++) o[u][j] = wmma_bf16(ap, bv[j], o[u][j]);
    }
  }

  int b = bh >> 4, h = bh & 15;
#pragma unroll
  for (int u = 0; u < 2; u++)
#pragma unroll
    for (int j = 0; j < 4; j++)
#pragma unroll
      for (int v = 0; v < 8; v++) {
        int row = q0 + u * 16 + v + half8;
        int d = j * 16 + ncol;
        float val = o[u][j][v] / lrow[u][v];
        ctx[(long)(b * SEQ + row) * CH + h * HD + d] = f2bf(val);
      }
}

// Output projection: ctx[8192,1024] @ w_out[1024,1024]^T -> fp32, 64x64/wave
__global__ void __launch_bounds__(128) out_gemm(const __bf16* __restrict__ A,
                                                const __bf16* __restrict__ W,
                                                float* __restrict__ out) {
  int wave = threadIdx.x >> 5;
  int lane = threadIdx.x & 31;
  int n0 = blockIdx.x * 256 + wave * 64;
  int m0 = blockIdx.y * 64;
  v8f acc[4][4] = {};
  for (int k0 = 0; k0 < CH; k0 += 32) {
    v16bf a[4], b[4];
#pragma unroll
    for (int i = 0; i < 4; i++) a[i] = load_frag(A, CH, m0 + i * 16, k0);
#pragma unroll
    for (int j = 0; j < 4; j++) b[j] = load_frag(W, CH, n0 + j * 16, k0);
#pragma unroll
    for (int i = 0; i < 4; i++)
#pragma unroll
      for (int j = 0; j < 4; j++)
        acc[i][j] = wmma_bf16(a[i], b[j], acc[i][j]);
  }
  int half8 = (lane & 16) ? 8 : 0;
  int ncol  = lane & 15;
#pragma unroll
  for (int i = 0; i < 4; i++)
#pragma unroll
    for (int j = 0; j < 4; j++)
#pragma unroll
      for (int v = 0; v < 8; v++) {
        int m = m0 + i * 16 + v + half8;
        int n = n0 + j * 16 + ncol;
        out[(long)m * CH + n] = acc[i][j][v];
      }
}

extern "C" void kernel_launch(void* const* d_in, const int* in_sizes, int n_in,
                              void* d_out, int out_size, void* d_ws, size_t ws_size,
                              hipStream_t stream) {
  const float* x     = (const float*)d_in[0];
  const float* w_in  = (const float*)d_in[1];
  const float* w_out = (const float*)d_in[2];
  float* out = (float*)d_out;

  const int M = BATCH * SEQ;   // 8192
  char* ws = (char*)d_ws;
  size_t off = 0;
  auto alloc = [&](size_t bytes) -> void* {
    void* p = ws + off;
    off += (bytes + 255) & ~(size_t)255;
    return p;
  };
  __bf16* xb   = (__bf16*)alloc((size_t)M * CH * 2);          // 16 MiB
  __bf16* wib  = (__bf16*)alloc((size_t)3 * CH * CH * 2);     // 6 MiB
  __bf16* wob  = (__bf16*)alloc((size_t)CH * CH * 2);         // 2 MiB
  __bf16* qb   = (__bf16*)alloc((size_t)M * CH * 2);          // 16 MiB
  __bf16* kb   = (__bf16*)alloc((size_t)M * CH * 2);          // 16 MiB
  __bf16* vtb  = (__bf16*)alloc((size_t)M * CH * 2);          // 16 MiB (V transposed)
  __bf16* ctxb = (__bf16*)alloc((size_t)M * CH * 2);          // 16 MiB

  convert_f32_bf16<<<2048, 256, 0, stream>>>(x,     xb,  M * CH);
  convert_f32_bf16<<<1024, 256, 0, stream>>>(w_in,  wib, 3 * CH * CH);
  convert_f32_bf16<<<512,  256, 0, stream>>>(w_out, wob, CH * CH);

  qkv_gemm<<<dim3(3 * CH / 256, M / 64), 128, 0, stream>>>(xb, wib, qb, kb, vtb);
  attn_fwd<<<dim3(SEQ / 32, BATCH * NHEADS), 32, 0, stream>>>(qb, kb, vtb, ctxb);
  out_gemm<<<dim3(CH / 256, M / 64), 128, 0, stream>>>(ctxb, wob, out);
}
